// GraphNetBlock_28114855920032
// MI455X (gfx1250) — matrix operand order repaired and checked by reading, not compiled, not run on hardware
//
#include <hip/hip_runtime.h>

typedef _Float16 __attribute__((ext_vector_type(16))) v16h;
typedef __fp16   __attribute__((ext_vector_type(2)))  pk2v;   // return type of cvt_pkrtz
typedef float    __attribute__((ext_vector_type(8)))  v8f;
typedef float    __attribute__((ext_vector_type(4)))  f4v;
typedef unsigned int u32;
typedef u32      __attribute__((ext_vector_type(4)))  u4v;

#define NN 50000
#define NE 600000
#define DD 128
#define LN_EPS 1e-5f

// f16 transposed-weight layout in workspace (half-element offsets)
#define W1E_OFF 0        // [128][384]  We1^T
#define W2E_OFF 49152    // [128][128]  We2^T
#define W1N_OFF 65536    // [128][256]  Wn1^T
#define W2N_OFF 98304    // [128][128]  Wn2^T
#define W_TOTAL 114688

union FragU { v16h v; u4v q[2]; };
union HU   { pk2v h; u32 u; };

__device__ __forceinline__ v8f wmma_f16(v16h a, v16h b, v8f c) {
  // D = A(16x32 f16) * B(32x16 f16) + C(16x16 f32)
  return __builtin_amdgcn_wmma_f32_16x16x32_f16(false, a, false, b, (short)0, c,
                                                false, false);
}

// ---- convert fp32 weights to f16, transposed so A-fragments are contiguous ----
__global__ void prep_weights_kernel(const float* __restrict__ We1,
                                    const float* __restrict__ We2,
                                    const float* __restrict__ Wn1,
                                    const float* __restrict__ Wn2,
                                    _Float16* __restrict__ w) {
  int idx = blockIdx.x * 256 + threadIdx.x;
  if (idx >= W_TOTAL) return;
  float v;
  if (idx < W2E_OFF)      { int j = idx / 384, k = idx % 384;          v = We1[k * 128 + j]; }
  else if (idx < W1N_OFF) { int t = idx - W2E_OFF; int o = t >> 7, j = t & 127; v = We2[j * 128 + o]; }
  else if (idx < W2N_OFF) { int t = idx - W1N_OFF; int j = t >> 8, k = t & 255; v = Wn1[k * 128 + j]; }
  else                    { int t = idx - W2N_OFF; int o = t >> 7, j = t & 127; v = Wn2[j * 128 + o]; }
  w[idx] = (_Float16)v;
}

// =================== edge update: MLP(concat(nf[s],nf[r],ef)) + LN, scatter ===================
__global__ void __launch_bounds__(256) edge_kernel(
    const float* __restrict__ nf, const float* __restrict__ ef,
    const int* __restrict__ snd, const int* __restrict__ rcv,
    const float* __restrict__ b1, const float* __restrict__ b2,
    const float* __restrict__ gam, const float* __restrict__ bet,
    const _Float16* __restrict__ wglob,
    float* agg, float* __restrict__ out_e) {
  extern __shared__ char smem[];
  _Float16* sW1 = (_Float16*)smem;        // [j 0..127][k 0..383]
  _Float16* sW2 = sW1 + 49152;            // [o 0..127][j 0..127]
  float* sB1 = (float*)(smem + 131072);
  float* sB2 = sB1 + 128;
  float* sG  = sB1 + 256;
  float* sBt = sB1 + 384;

  const int tid = threadIdx.x;
  {
    const u4v* src = (const u4v*)wglob;
    u4v* dst = (u4v*)smem;
    for (int i = tid; i < 8192; i += 256) dst[i] = src[i];   // 128KB weights -> LDS
    if (tid < 128) { sB1[tid] = b1[tid]; sB2[tid] = b2[tid]; sG[tid] = gam[tid]; sBt[tid] = bet[tid]; }
  }
  __syncthreads();

  const int lane = tid & 31;
  const int wave = tid >> 5;
  const int col  = lane & 15;   // B-matrix column (edge) / A-matrix row select
  const int hi   = lane >> 4;
  const int off8 = hi * 8;      // accumulator row offset (D rows 8*hi..8*hi+7)
  const int klo  = hi * 16;     // B-fragment K base

  for (int tile = blockIdx.x * 8 + wave; tile < NE / 16; tile += gridDim.x * 8) {
    const int e = tile * 16 + col;
    const int s = snd[e];
    const int r = rcv[e];
    const float* xs = nf + (size_t)s * DD;
    const float* xr = nf + (size_t)r * DD;
    const float* xe = ef + (size_t)e * DD;

    // ---- GEMM1: h^T[j, e] = sum_k W1[k,j] * x[e,k],  K = 384 ----
    v8f acc1[8] = {};
    for (int kt = 0; kt < 12; ++kt) {
      const float* src = (kt < 4) ? xs : (kt < 8) ? xr : xe;
      const float* p = src + ((kt & 3) * 32 + klo);
      f4v f0 = ((const f4v*)p)[0];
      f4v f1 = ((const f4v*)p)[1];
      f4v f2 = ((const f4v*)p)[2];
      f4v f3 = ((const f4v*)p)[3];
      FragU bf;
#pragma unroll
      for (int i = 0; i < 4; ++i) {
        bf.v[i]      = (_Float16)f0[i];
        bf.v[4 + i]  = (_Float16)f1[i];
        bf.v[8 + i]  = (_Float16)f2[i];
        bf.v[12 + i] = (_Float16)f3[i];
      }
      const _Float16* wrow = sW1 + kt * 32 + klo;
#pragma unroll
      for (int jt = 0; jt < 8; ++jt) {
        const _Float16* wp = wrow + (jt * 16 + col) * 384;
        FragU af;
        af.q[0] = *(const u4v*)wp;
        af.q[1] = *(const u4v*)(wp + 8);
        acc1[jt] = wmma_f16(af.v, bf.v, acc1[jt]);
      }
    }

    // ---- bias + ReLU, pack to f16 pairs ----
    u4v pk[8];
#pragma unroll
    for (int jt = 0; jt < 8; ++jt) {
      const float* bp = sB1 + jt * 16 + off8;
#pragma unroll
      for (int v = 0; v < 8; ++v) acc1[jt][v] = fmaxf(acc1[jt][v] + bp[v], 0.f);
#pragma unroll
      for (int i = 0; i < 4; ++i) {
        HU hu; hu.h = __builtin_amdgcn_cvt_pkrtz(acc1[jt][2 * i], acc1[jt][2 * i + 1]);
        pk[jt][i] = hu.u;
      }
    }
    // ---- C-layout -> B-fragment: single half-wave exchange per 32-K block ----
    FragU bf2[4];
#pragma unroll
    for (int jb = 0; jb < 4; ++jb) {
      u4v own0 = pk[2 * jb], own1 = pk[2 * jb + 1];
      u4v sw0, sw1;
#pragma unroll
      for (int i = 0; i < 4; ++i) {
        sw0[i] = (u32)__shfl_xor((int)own0[i], 16, 32);
        sw1[i] = (u32)__shfl_xor((int)own1[i], 16, 32);
      }
      bf2[jb].q[0] = hi ? sw1 : own0;
      bf2[jb].q[1] = hi ? own1 : sw0;
    }

    // ---- GEMM2: y^T[o, e] = sum_j W2[j,o] * h[e,j],  K = 128 ----
    v8f acc2[8] = {};
#pragma unroll
    for (int jb = 0; jb < 4; ++jb) {
      const _Float16* wrow = sW2 + jb * 32 + klo;
#pragma unroll
      for (int ot = 0; ot < 8; ++ot) {
        const _Float16* wp = wrow + (ot * 16 + col) * 128;
        FragU af;
        af.q[0] = *(const u4v*)wp;
        af.q[1] = *(const u4v*)(wp + 8);
        acc2[ot] = wmma_f16(af.v, bf2[jb].v, acc2[ot]);
      }
    }

    // ---- epilogue: +b2, LayerNorm across 128 (2 lanes per edge), scatter, residual ----
    float s1 = 0.f, s2 = 0.f;
#pragma unroll
    for (int ot = 0; ot < 8; ++ot) {
      const float* bp = sB2 + ot * 16 + off8;
#pragma unroll
      for (int v = 0; v < 8; ++v) {
        float t = acc2[ot][v] + bp[v];
        acc2[ot][v] = t;
        s1 += t;
        s2 += t * t;
      }
    }
    s1 += __shfl_xor(s1, 16, 32);
    s2 += __shfl_xor(s2, 16, 32);
    const float mu = s1 * (1.f / 128.f);
    const float var = s2 * (1.f / 128.f) - mu * mu;
    const float rinv = rsqrtf(var + LN_EPS);

    float* ap = agg + (size_t)r * DD;
    float* op = out_e + (size_t)e * DD;
    const float* rp = ef + (size_t)e * DD;
#pragma unroll
    for (int ot = 0; ot < 8; ++ot) {
      const int ob = ot * 16 + off8;
      const float* gp = sG + ob;
      const float* bp = sBt + ob;
      f4v r0 = ((const f4v*)(rp + ob))[0];
      f4v r1 = ((const f4v*)(rp + ob))[1];
      float ln[8];
#pragma unroll
      for (int v = 0; v < 8; ++v) ln[v] = (acc2[ot][v] - mu) * rinv * gp[v] + bp[v];
#pragma unroll
      for (int v = 0; v < 8; ++v) atomicAdd(ap + ob + v, ln[v]);   // agg uses pre-residual
      f4v o0 = { ln[0] + r0[0], ln[1] + r0[1], ln[2] + r0[2], ln[3] + r0[3] };
      f4v o1 = { ln[4] + r1[0], ln[5] + r1[1], ln[6] + r1[2], ln[7] + r1[3] };
      ((f4v*)(op + ob))[0] = o0;
      ((f4v*)(op + ob))[1] = o1;
    }
  }
}

// =================== node update: MLP(concat(nf, agg)) + LN + residual (in-place over agg) ===================
__global__ void __launch_bounds__(256) node_kernel(
    const float* __restrict__ nf,
    const float* __restrict__ b1, const float* __restrict__ b2,
    const float* __restrict__ gam, const float* __restrict__ bet,
    const _Float16* __restrict__ wglob,   // points at Wn1^T
    const float* agg, float* out_n) {
  extern __shared__ char smem[];
  _Float16* sW1 = (_Float16*)smem;        // [j 0..127][k 0..255]
  _Float16* sW2 = sW1 + 32768;            // [o 0..127][j 0..127]
  float* sB1 = (float*)(smem + 98304);
  float* sB2 = sB1 + 128;
  float* sG  = sB1 + 256;
  float* sBt = sB1 + 384;

  const int tid = threadIdx.x;
  {
    const u4v* src = (const u4v*)wglob;
    u4v* dst = (u4v*)smem;
    for (int i = tid; i < 6144; i += 256) dst[i] = src[i];   // 96KB weights -> LDS
    if (tid < 128) { sB1[tid] = b1[tid]; sB2[tid] = b2[tid]; sG[tid] = gam[tid]; sBt[tid] = bet[tid]; }
  }
  __syncthreads();

  const int lane = tid & 31;
  const int wave = tid >> 5;
  const int col  = lane & 15;
  const int hi   = lane >> 4;
  const int off8 = hi * 8;
  const int klo  = hi * 16;

  for (int tile = blockIdx.x * 8 + wave; tile < NN / 16; tile += gridDim.x * 8) {
    const int vtx = tile * 16 + col;
    const float* xn = nf + (size_t)vtx * DD;
    const float* xa = agg + (size_t)vtx * DD;

    v8f acc1[8] = {};
    for (int kt = 0; kt < 8; ++kt) {            // K = 256
      const float* src = (kt < 4) ? xn : xa;
      const float* p = src + ((kt & 3) * 32 + klo);
      f4v f0 = ((const f4v*)p)[0];
      f4v f1 = ((const f4v*)p)[1];
      f4v f2 = ((const f4v*)p)[2];
      f4v f3 = ((const f4v*)p)[3];
      FragU bf;
#pragma unroll
      for (int i = 0; i < 4; ++i) {
        bf.v[i]      = (_Float16)f0[i];
        bf.v[4 + i]  = (_Float16)f1[i];
        bf.v[8 + i]  = (_Float16)f2[i];
        bf.v[12 + i] = (_Float16)f3[i];
      }
      const _Float16* wrow = sW1 + kt * 32 + klo;
#pragma unroll
      for (int jt = 0; jt < 8; ++jt) {
        const _Float16* wp = wrow + (jt * 16 + col) * 256;
        FragU af;
        af.q[0] = *(const u4v*)wp;
        af.q[1] = *(const u4v*)(wp + 8);
        acc1[jt] = wmma_f16(af.v, bf.v, acc1[jt]);
      }
    }

    u4v pk[8];
#pragma unroll
    for (int jt = 0; jt < 8; ++jt) {
      const float* bp = sB1 + jt * 16 + off8;
#pragma unroll
      for (int v = 0; v < 8; ++v) acc1[jt][v] = fmaxf(acc1[jt][v] + bp[v], 0.f);
#pragma unroll
      for (int i = 0; i < 4; ++i) {
        HU hu; hu.h = __builtin_amdgcn_cvt_pkrtz(acc1[jt][2 * i], acc1[jt][2 * i + 1]);
        pk[jt][i] = hu.u;
      }
    }
    FragU bf2[4];
#pragma unroll
    for (int jb = 0; jb < 4; ++jb) {
      u4v own0 = pk[2 * jb], own1 = pk[2 * jb + 1];
      u4v sw0, sw1;
#pragma unroll
      for (int i = 0; i < 4; ++i) {
        sw0[i] = (u32)__shfl_xor((int)own0[i], 16, 32);
        sw1[i] = (u32)__shfl_xor((int)own1[i], 16, 32);
      }
      bf2[jb].q[0] = hi ? sw1 : own0;
      bf2[jb].q[1] = hi ? own1 : sw0;
    }

    v8f acc2[8] = {};
#pragma unroll
    for (int jb = 0; jb < 4; ++jb) {
      const _Float16* wrow = sW2 + jb * 32 + klo;
#pragma unroll
      for (int ot = 0; ot < 8; ++ot) {
        const _Float16* wp = wrow + (ot * 16 + col) * 128;
        FragU af;
        af.q[0] = *(const u4v*)wp;
        af.q[1] = *(const u4v*)(wp + 8);
        acc2[ot] = wmma_f16(af.v, bf2[jb].v, acc2[ot]);
      }
    }

    float s1 = 0.f, s2 = 0.f;
#pragma unroll
    for (int ot = 0; ot < 8; ++ot) {
      const float* bp = sB2 + ot * 16 + off8;
#pragma unroll
      for (int v = 0; v < 8; ++v) {
        float t = acc2[ot][v] + bp[v];
        acc2[ot][v] = t;
        s1 += t;
        s2 += t * t;
      }
    }
    s1 += __shfl_xor(s1, 16, 32);
    s2 += __shfl_xor(s2, 16, 32);
    const float mu = s1 * (1.f / 128.f);
    const float var = s2 * (1.f / 128.f) - mu * mu;
    const float rinv = rsqrtf(var + LN_EPS);

    float* op = out_n + (size_t)vtx * DD;
#pragma unroll
    for (int ot = 0; ot < 8; ++ot) {
      const int ob = ot * 16 + off8;
      const float* gp = sG + ob;
      const float* bp = sBt + ob;
      f4v r0 = ((const f4v*)(xn + ob))[0];
      f4v r1 = ((const f4v*)(xn + ob))[1];
      f4v o0, o1;
#pragma unroll
      for (int v = 0; v < 4; ++v) {
        o0[v] = (acc2[ot][v] - mu) * rinv * gp[v] + bp[v] + r0[v];
        o1[v] = (acc2[ot][4 + v] - mu) * rinv * gp[4 + v] + bp[4 + v] + r1[v];
      }
      ((f4v*)(op + ob))[0] = o0;
      ((f4v*)(op + ob))[1] = o1;
    }
  }
}

extern "C" void kernel_launch(void* const* d_in, const int* in_sizes, int n_in,
                              void* d_out, int out_size, void* d_ws, size_t ws_size,
                              hipStream_t stream) {
  const float* nf  = (const float*)d_in[0];
  const float* ef  = (const float*)d_in[1];
  const int*   snd = (const int*)d_in[2];
  const int*   rcv = (const int*)d_in[3];
  const float* We1 = (const float*)d_in[4];
  const float* be1 = (const float*)d_in[5];
  const float* We2 = (const float*)d_in[6];
  const float* be2 = (const float*)d_in[7];
  const float* ge  = (const float*)d_in[8];
  const float* bbe = (const float*)d_in[9];
  const float* Wn1 = (const float*)d_in[10];
  const float* bn1 = (const float*)d_in[11];
  const float* Wn2 = (const float*)d_in[12];
  const float* bn2 = (const float*)d_in[13];
  const float* gn  = (const float*)d_in[14];
  const float* bbn = (const float*)d_in[15];

  float* out_nodes = (float*)d_out;                       // doubles as the agg accumulator
  float* out_edges = out_nodes + (size_t)NN * DD;
  _Float16* w16 = (_Float16*)d_ws;                        // 229 KB of f16 weights

  (void)hipMemsetAsync(out_nodes, 0, (size_t)NN * DD * sizeof(float), stream);
  prep_weights_kernel<<<(W_TOTAL + 255) / 256, 256, 0, stream>>>(We1, We2, Wn1, Wn2, w16);
  edge_kernel<<<1024, 256, 133120, stream>>>(nf, ef, snd, rcv, be1, be2, ge, bbe,
                                             w16, out_nodes, out_edges);
  node_kernel<<<392, 256, 100352, stream>>>(nf, bn1, bn2, gn, bbn,
                                            w16 + W1N_OFF, out_nodes, out_nodes);
}
